// GraphMultiHeadSelfAttention_59150289600768
// MI455X (gfx1250) — matrix operand order repaired and checked by reading, not compiled, not run on hardware
//
#include <hip/hip_runtime.h>
#include <hip/hip_bf16.h>

#define B_ 2
#define N_ 4096
#define E_ 256
#define H_ 8
#define HS_ 32

typedef __attribute__((ext_vector_type(16))) __bf16 bf16x16;
typedef __attribute__((ext_vector_type(8)))  float  f32x8;

union Frag {
  unsigned int u[8];
  uint4 q[2];
  bf16x16 v;
};

// pack two f32 -> packed bf16 pair (low = a, high = b)
__device__ __forceinline__ unsigned int pk_bf16(float a, float b) {
#if __has_builtin(__builtin_amdgcn_cvt_pk_bf16_f32)
  return __builtin_bit_cast(unsigned int, __builtin_amdgcn_cvt_pk_bf16_f32(a, b));
#else
  // round-half-up then merge high halves with one v_perm_b32
  unsigned int ua = __float_as_uint(a) + 0x8000u;
  unsigned int ub = __float_as_uint(b) + 0x8000u;
  return __builtin_amdgcn_perm(ub, ua, 0x07060302u);
#endif
}
__device__ __forceinline__ unsigned short to_bf16(float a) {
  return (unsigned short)((__float_as_uint(a) + 0x8000u) >> 16);
}
__device__ __forceinline__ float fast_exp2(float x) {
#if __has_builtin(__builtin_amdgcn_exp2f)
  return __builtin_amdgcn_exp2f(x);   // v_exp_f32
#else
  return exp2f(x);
#endif
}
// xor-16 lane swap (group-of-32 ds_swizzle: xor=0x10, and=0x1f)
__device__ __forceinline__ float swz16f(float x) {
  return __int_as_float(__builtin_amdgcn_ds_swizzle(__float_as_int(x), 0x401F));
}
__device__ __forceinline__ unsigned int swz16u(unsigned int x) {
  return (unsigned int)__builtin_amdgcn_ds_swizzle((int)x, 0x401F);
}
__device__ __forceinline__ float bperm_f(int srclane, float x) {
  return __int_as_float(__builtin_amdgcn_ds_bpermute(srclane << 2, __float_as_int(x)));
}

// ---------------------------------------------------------------------------
// Projection: out[b][h][n][hs] (bf16) = (X @ W^T + bias) * scale  (Q and K)
// Wave tile: 16 rows (n) x 16 cols (e_out), K-loop over 256 in steps of 32.
// For Q, scale = 1/sqrt(HS)*log2(e) folds the softmax scale into the GEMM
// epilogue so the attention hot loop needs no per-score multiply.
// ---------------------------------------------------------------------------
__global__ void __launch_bounds__(256)
proj_rowmajor(const float* __restrict__ X, const float* __restrict__ W,
              const float* __restrict__ bias, unsigned short* __restrict__ out,
              float scale) {
  const int lane = threadIdx.x & 31;
  const int lq   = lane & 15;
  const int half = lane >> 4;
  const int waveId = blockIdx.x * 8 + (threadIdx.x >> 5);   // 0..8191
  const int i0 = (waveId >> 4) * 16;                        // global row (b*N+n)
  const int e0 = (waveId & 15) * 16;                        // output feature

  const float4* xr = (const float4*)(X + (size_t)(i0 + lq) * E_);  // A: X rows
  const float4* wr = (const float4*)(W + (size_t)(e0 + lq) * E_);  // B: W rows (= W^T cols)
  f32x8 c = {0.f,0.f,0.f,0.f,0.f,0.f,0.f,0.f};
#pragma unroll
  for (int kk = 0; kk < E_ / 32; ++kk) {
    const int kb4 = kk * 8;
    Frag a, b;
    // A-fragment (16x32): lane<16 -> k 0..7 & 16..23 ; lane>=16 -> k 8..15 & 24..31
    float4 a0 = xr[kb4 + 2*half + 0];
    float4 a1 = xr[kb4 + 2*half + 1];
    float4 a2 = xr[kb4 + 2*half + 4];
    float4 a3 = xr[kb4 + 2*half + 5];
    a.u[0]=pk_bf16(a0.x,a0.y); a.u[1]=pk_bf16(a0.z,a0.w);
    a.u[2]=pk_bf16(a1.x,a1.y); a.u[3]=pk_bf16(a1.z,a1.w);
    a.u[4]=pk_bf16(a2.x,a2.y); a.u[5]=pk_bf16(a2.z,a2.w);
    a.u[6]=pk_bf16(a3.x,a3.y); a.u[7]=pk_bf16(a3.z,a3.w);
    // B-fragment (32x16): lane<16 -> k 0..7 & 16..23 ; lane>=16 -> k 8..15 & 24..31
    float4 b0 = wr[kb4 + 2*half + 0];
    float4 b1 = wr[kb4 + 2*half + 1];
    float4 b2 = wr[kb4 + 2*half + 4];
    float4 b3 = wr[kb4 + 2*half + 5];
    b.u[0]=pk_bf16(b0.x,b0.y); b.u[1]=pk_bf16(b0.z,b0.w);
    b.u[2]=pk_bf16(b1.x,b1.y); b.u[3]=pk_bf16(b1.z,b1.w);
    b.u[4]=pk_bf16(b2.x,b2.y); b.u[5]=pk_bf16(b2.z,b2.w);
    b.u[6]=pk_bf16(b3.x,b3.y); b.u[7]=pk_bf16(b3.z,b3.w);
    c = __builtin_amdgcn_wmma_f32_16x16x32_bf16(false, a.v, false, b.v,
                                                (short)0, c, false, false);
  }
  const float bbs = bias[e0 + lq] * scale;   // uniform-per-lane epilogue
  const int h   = e0 >> 5;
  const int hs0 = e0 & 31;
#pragma unroll
  for (int r = 0; r < 8; ++r) {
    int ng = i0 + r + 8 * half;       // global row
    int bI = ng >> 12;
    int n  = ng & (N_ - 1);
    out[((size_t)(bI * H_ + h) * N_ + n) * HS_ + hs0 + lq] =
        to_bf16(fmaf(c[r], scale, bbs));
  }
}

// ---------------------------------------------------------------------------
// Projection producing V^T directly: out[b][h][hs][n] (bf16) = (X@Wv^T+bv)^T
// Computed as Wv @ X^T so stores are coalesced along n.
// ---------------------------------------------------------------------------
__global__ void __launch_bounds__(256)
proj_vt(const float* __restrict__ X, const float* __restrict__ W,
        const float* __restrict__ bias, unsigned short* __restrict__ out) {
  const int lane = threadIdx.x & 31;
  const int lq   = lane & 15;
  const int half = lane >> 4;
  const int waveId = blockIdx.x * 8 + (threadIdx.x >> 5);   // 0..8191
  const int e0 = (waveId >> 9) * 16;                        // e_out tile (0..255)
  const int n0 = (waveId & 511) * 16;                       // global row tile

  const float4* wr = (const float4*)(W + (size_t)(e0 + lq) * E_);  // A: Wv rows
  const float4* xr = (const float4*)(X + (size_t)(n0 + lq) * E_);  // B: X rows (= X^T cols)
  f32x8 c = {0.f,0.f,0.f,0.f,0.f,0.f,0.f,0.f};
#pragma unroll
  for (int kk = 0; kk < E_ / 32; ++kk) {
    const int kb4 = kk * 8;
    Frag a, b;
    float4 a0 = wr[kb4 + 2*half + 0];
    float4 a1 = wr[kb4 + 2*half + 1];
    float4 a2 = wr[kb4 + 2*half + 4];
    float4 a3 = wr[kb4 + 2*half + 5];
    a.u[0]=pk_bf16(a0.x,a0.y); a.u[1]=pk_bf16(a0.z,a0.w);
    a.u[2]=pk_bf16(a1.x,a1.y); a.u[3]=pk_bf16(a1.z,a1.w);
    a.u[4]=pk_bf16(a2.x,a2.y); a.u[5]=pk_bf16(a2.z,a2.w);
    a.u[6]=pk_bf16(a3.x,a3.y); a.u[7]=pk_bf16(a3.z,a3.w);
    float4 b0 = xr[kb4 + 2*half + 0];
    float4 b1 = xr[kb4 + 2*half + 1];
    float4 b2 = xr[kb4 + 2*half + 4];
    float4 b3 = xr[kb4 + 2*half + 5];
    b.u[0]=pk_bf16(b0.x,b0.y); b.u[1]=pk_bf16(b0.z,b0.w);
    b.u[2]=pk_bf16(b1.x,b1.y); b.u[3]=pk_bf16(b1.z,b1.w);
    b.u[4]=pk_bf16(b2.x,b2.y); b.u[5]=pk_bf16(b2.z,b2.w);
    b.u[6]=pk_bf16(b3.x,b3.y); b.u[7]=pk_bf16(b3.z,b3.w);
    c = __builtin_amdgcn_wmma_f32_16x16x32_bf16(false, a.v, false, b.v,
                                                (short)0, c, false, false);
  }
#pragma unroll
  for (int r = 0; r < 8; ++r) {
    int eo = e0 + r + 8 * half;
    int h  = eo >> 5;
    int hs = eo & 31;
    int ng = n0 + lq;
    int bI = ng >> 12;
    int n  = ng & (N_ - 1);
    out[((size_t)(bI * H_ + h) * HS_ + hs) * N_ + n] = to_bf16(c[r] + bias[eo]);
  }
}

// ---------------------------------------------------------------------------
// Flash attention: per wave one (b, h, 16-query tile); stream keys in 32-chunks.
// Computes S^T = K_chunk @ Q^T so probabilities land in the A-operand layout
// needed for O = P @ V with zero cross-lane transposes. Softmax in log2 domain
// (scale pre-folded into Q) so exponentials are raw v_exp_f32.
// ---------------------------------------------------------------------------
__global__ void __launch_bounds__(256)
attn_fa(const unsigned short* __restrict__ qb, const unsigned short* __restrict__ kb,
        const unsigned short* __restrict__ vtb, const unsigned char* __restrict__ adj,
        float* __restrict__ out) {
  const int lane = threadIdx.x & 31;
  const int lq   = lane & 15;
  const int half = lane >> 4;
  const bool lt16 = (half == 0);
  const int waveId = blockIdx.x * 8 + (threadIdx.x >> 5);   // 0..4095
  const int i0 = (waveId & 255) * 16;
  const int h  = (waveId >> 8) & 7;
  const int b  = waveId >> 11;

  const unsigned short* qh = qb  + (size_t)(b * H_ + h) * N_ * HS_;
  const unsigned short* kh = kb  + (size_t)(b * H_ + h) * N_ * HS_;
  const unsigned short* vh = vtb + (size_t)(b * H_ + h) * HS_ * N_;
  const unsigned char*  ar = adj + (size_t)b * N_ * N_ + (size_t)(i0 + lq) * N_ + 16 * half;

  // Q^T B-fragment: lane q -> col q (query row q), k(hs) 8h..8h+7 & 16+8h..16+8h+7 by half h
  Frag qf;
  {
    const uint4* qr = (const uint4*)(qh + (size_t)(i0 + lq) * HS_);
    qf.q[0] = qr[half];
    qf.q[1] = qr[half + 2];
  }

  f32x8 acc0 = {0.f,0.f,0.f,0.f,0.f,0.f,0.f,0.f};
  f32x8 acc1 = {0.f,0.f,0.f,0.f,0.f,0.f,0.f,0.f};
  float m_run = -3.0e38f;
  float l_run = 0.f;
  const float NEG = -3.0e38f;

#pragma unroll 2
  for (int j = 0; j < N_; j += 32) {
    // K A-fragments (16 keys x 32 hs each)
    Frag ak0, ak1, bv0, bv1;
    const uint4* k0p = (const uint4*)(kh + (size_t)(j + lq) * HS_);
    ak0.q[0] = k0p[half];  ak0.q[1] = k0p[half + 2];
    const uint4* k1p = (const uint4*)(kh + (size_t)(j + 16 + lq) * HS_);
    ak1.q[0] = k1p[half];  ak1.q[1] = k1p[half + 2];
    // V B-fragments from V^T: lane c -> hs col, keys j+8*half..+7 and j+16+8*half..+7
    const uint4* v0p = (const uint4*)(vh + (size_t)lq * N_ + j + 8 * half);
    bv0.q[0] = v0p[0];  bv0.q[1] = v0p[2];
    const uint4* v1p = (const uint4*)(vh + (size_t)(16 + lq) * N_ + j + 8 * half);
    bv1.q[0] = v1p[0];  bv1.q[1] = v1p[2];
    // adj: lane L holds row (i0+L%16), cols j+16*(L/16)..+15
    uint4 av = *(const uint4*)(ar + j);

    f32x8 zero = {0.f,0.f,0.f,0.f,0.f,0.f,0.f,0.f};
    f32x8 s0 = __builtin_amdgcn_wmma_f32_16x16x32_bf16(false, ak0.v, false, qf.v,
                                                       (short)0, zero, false, false);
    f32x8 s1 = __builtin_amdgcn_wmma_f32_16x16x32_bf16(false, ak1.v, false, qf.v,
                                                       (short)0, zero, false, false);

    // redistribute adj bytes: lane q needs keys of its half; swap 8B halves q<->q+16
    unsigned int slo0 = swz16u(av.x), slo1 = swz16u(av.y);
    unsigned int shi0 = swz16u(av.z), shi1 = swz16u(av.w);
    unsigned int d00 = lt16 ? av.x : shi0;   // S^T0 keys: bytes r
    unsigned int d01 = lt16 ? av.y : shi1;
    unsigned int d10 = lt16 ? slo0 : av.z;   // S^T1 keys
    unsigned int d11 = lt16 ? slo1 : av.w;

    float mx = NEG;
#pragma unroll
    for (int r = 0; r < 8; ++r) {
      unsigned int m0 = ((r < 4 ? (d00 >> (8 * r)) : (d01 >> (8 * (r - 4)))) & 0xFFu);
      unsigned int m1 = ((r < 4 ? (d10 >> (8 * r)) : (d11 >> (8 * (r - 4)))) & 0xFFu);
      s0[r] = m0 ? s0[r] : NEG;     // scale pre-folded into Q projection
      s1[r] = m1 ? s1[r] : NEG;
      mx = fmaxf(mx, fmaxf(s0[r], s1[r]));
    }
    mx = fmaxf(mx, swz16f(mx));            // combine lane q / q+16 halves
    float m_new = fmaxf(m_run, mx);
    float alpha = fast_exp2(m_run - m_new);
    m_run = m_new;

    float sm = 0.f;
#pragma unroll
    for (int r = 0; r < 8; ++r) {
      s0[r] = fast_exp2(s0[r] - m_new);
      s1[r] = fast_exp2(s1[r] - m_new);
      sm += s0[r] + s1[r];
    }
    sm += swz16f(sm);
    l_run = l_run * alpha + sm;

    // P in A-operand layout: lane-aligned with S^T C/D layout -> just pack pairs
    Frag pf;
    pf.u[0] = pk_bf16(s0[0], s0[1]); pf.u[1] = pk_bf16(s0[2], s0[3]);
    pf.u[2] = pk_bf16(s0[4], s0[5]); pf.u[3] = pk_bf16(s0[6], s0[7]);
    pf.u[4] = pk_bf16(s1[0], s1[1]); pf.u[5] = pk_bf16(s1[2], s1[3]);
    pf.u[6] = pk_bf16(s1[4], s1[5]); pf.u[7] = pk_bf16(s1[6], s1[7]);

    // rescale O accumulators: query of acc reg r is r+8*(lane>=16); alpha[q] lives
    // in lanes q and q+16 -> broadcast with ds_bpermute
#pragma unroll
    for (int r = 0; r < 8; ++r) {
      float a = bperm_f(r + (lt16 ? 0 : 24), alpha);
      acc0[r] *= a;
      acc1[r] *= a;
    }
    acc0 = __builtin_amdgcn_wmma_f32_16x16x32_bf16(false, pf.v, false, bv0.v,
                                                   (short)0, acc0, false, false);
    acc1 = __builtin_amdgcn_wmma_f32_16x16x32_bf16(false, pf.v, false, bv1.v,
                                                   (short)0, acc1, false, false);
  }

  float linv = 1.0f / l_run;
#pragma unroll
  for (int r = 0; r < 8; ++r) {
    float li = bperm_f(r + (lt16 ? 0 : 24), linv);
    int nrow = i0 + r + 8 * half;
    float* op = out + (size_t)(b * N_ + nrow) * E_ + h * HS_ + lq;
    op[0]  = acc0[r] * li;    // hs cols 0..15
    op[16] = acc1[r] * li;    // hs cols 16..31
  }
}

// ---------------------------------------------------------------------------
extern "C" void kernel_launch(void* const* d_in, const int* in_sizes, int n_in,
                              void* d_out, int out_size, void* d_ws, size_t ws_size,
                              hipStream_t stream) {
  const float*         X   = (const float*)d_in[0];
  const unsigned char* adj = (const unsigned char*)d_in[1];   // bool bytes
  const float* Wq = (const float*)d_in[2];
  const float* bq = (const float*)d_in[3];
  const float* Wk = (const float*)d_in[4];
  const float* bk = (const float*)d_in[5];
  const float* Wv = (const float*)d_in[6];
  const float* bv = (const float*)d_in[7];

  // softmax scale (1/sqrt(HS)) * log2(e), folded into Q
  const float QSCALE = 0.17677669529663688f * 1.4426950408889634f;

  const size_t perMat = (size_t)B_ * H_ * N_ * HS_;   // 2,097,152 bf16 elems = 4MB
  unsigned short* qb  = (unsigned short*)d_ws;
  unsigned short* kb  = qb + perMat;
  unsigned short* vtb = kb + perMat;

  proj_rowmajor<<<1024, 256, 0, stream>>>(X, Wq, bq, qb, QSCALE);
  proj_rowmajor<<<1024, 256, 0, stream>>>(X, Wk, bk, kb, 1.0f);
  proj_vt      <<<1024, 256, 0, stream>>>(X, Wv, bv, vtb);
  attn_fa      <<< 512, 256, 0, stream>>>(qb, kb, vtb, adj, (float*)d_out);
}
